// MultiHeadAttention_43078521979665
// MI455X (gfx1250) — compile-verified, matrix operand order checked by reading, not run on hardware
//
#include <hip/hip_runtime.h>
#include <hip/hip_bf16.h>

// ---------------------------------------------------------------------------
// Multi-head attention forward for MI455X (gfx1250, wave32, WMMA + TDM).
//
// All matmuls run v_wmma_f32_16x16x32_bf16 (f32 accumulate). All GEMM operand
// tiles that are plain 2D bf16 copies are fetched with the Tensor Data Mover
// (tensor_load_to_lds, TENSORcnt double-buffering, hardware LDS padding to a
// bank-conflict-free stride). Only the attn-probability tile (f32 -> bf16)
// needs VALU staging, done with float4 loads + packed converts + b128 stores.
//
// Pipeline:
//   0) cvt q/k/v -> bf16 ; transpose Wq/Wk/Wv/Wo -> bf16 N-major   (ws)
//   1) gemm_bf16 x3 : X @ W + b -> Qh/Kh head-split bf16, Vh^T bf16 (TDM+WMMA)
//   2) attn_logits  : 0.125 * Qh Kh^T + mask -> attn f32 (d_out)    (TDM+WMMA)
//   3) softmax_rows : in-place row softmax                          (shfl)
//   4) attn_v       : attn @ Vh -> ctx bf16 merged                  (TDM+WMMA)
//   5) gemm_bf16    : ctx @ Wo + bo -> out f32 (d_out)              (TDM+WMMA)
// ---------------------------------------------------------------------------

typedef __bf16        v16bf __attribute__((ext_vector_type(16)));
typedef __bf16        v8bf  __attribute__((ext_vector_type(8)));
typedef float         v8f   __attribute__((ext_vector_type(8)));
typedef unsigned int  u32x4 __attribute__((ext_vector_type(4)));
typedef int           i32x4 __attribute__((ext_vector_type(4)));
typedef int           i32x8 __attribute__((ext_vector_type(8)));

#define S_LEN   2048
#define D_MODEL 1024
#define NHEAD   16
#define DEPTH   64
#define BATCH   2
#define MROWS   (BATCH * S_LEN)   // 4096
#define LDL     40                // LDS stride for 32-wide tiles: 80B (16B mult.)
#define LDQ     72                // LDS stride for 64-wide tiles: 144B

// Low 32 bits of a generic pointer into LDS == byte offset in LDS.
__device__ inline unsigned lds_off(const void* p) {
    return (unsigned)(unsigned long long)p;
}

// ---------------------------------------------------------------------------
// TDM: issue a 2D bf16 tile load global -> LDS (async, TENSORcnt-tracked).
//   tile_w elems wide (row), tile_h rows, row stride in elems, with LDS
//   padding: pad_interval (0=2,1=4,2=8,3=16,4=32.. DWORDs) / pad_amount
//   (value+1 DWORDs) so the LDS row stride becomes tile_w*2 + pad bytes.
// Descriptor packing per CDNA5 ISA D# groups 0/1 (data_size=1 -> 2 bytes).
// ---------------------------------------------------------------------------
__device__ inline void tdm_load_2d(unsigned lds_addr, const void* gptr,
                                   unsigned tile_w, unsigned tile_h,
                                   unsigned row_stride,
                                   unsigned pad_interval, unsigned pad_amount) {
    const unsigned long long ga = (unsigned long long)(size_t)gptr;
    u32x4 g0;
    g0[0] = 1u;                                           // count=1 (valid)
    g0[1] = lds_addr;                                     // lds_addr [63:32]
    g0[2] = (unsigned)ga;                                 // global_addr[31:0]
    g0[3] = (unsigned)((ga >> 32) & 0x1FFFFFFu) | (2u << 30); // addr[56:32]|type=2
    i32x8 g1;
    g1[0] = (int)((1u << 16)                 // data_size = 2 bytes
                | (1u << 20)                 // pad_enable
                | (pad_interval << 22)
                | (pad_amount << 25));
    g1[1] = (int)((tile_w & 0xFFFFu) << 16);              // tensor_dim0[15:0]
    g1[2] = (int)(((tile_w >> 16) & 0xFFFFu)              // tensor_dim0[31:16]
                | ((tile_h & 0xFFFFu) << 16));            // tensor_dim1[15:0]
    g1[3] = (int)(((tile_h >> 16) & 0xFFFFu)              // tensor_dim1[31:16]
                | ((tile_w & 0xFFFFu) << 16));            // tile_dim0
    g1[4] = (int)(tile_h & 0xFFFFu);                      // tile_dim1 (tile_dim2=0)
    g1[5] = (int)row_stride;                              // tensor_dim0_stride[31:0]
    g1[6] = 0;
    g1[7] = 0;
    const i32x4 z4 = {0, 0, 0, 0};
#if __clang_major__ >= 23
    const i32x8 z8 = {0, 0, 0, 0, 0, 0, 0, 0};
    __builtin_amdgcn_tensor_load_to_lds(g0, g1, z4, z4, z8, 0);
#else
    __builtin_amdgcn_tensor_load_to_lds(g0, g1, z4, z4, 0);
#endif
}

// Load one 16x32 bf16 WMMA operand fragment from a row-major LDS tile
// (A from M-major tile, B from the N-major / transposed tile).
__device__ inline v16bf frag16(const __bf16* p, int lane, int ld) {
    const int r  = lane & 15;
    const int h8 = (lane >> 4) << 3;           // 0 or 8
    const v8bf lo = *(const v8bf*)(p + r * ld + h8);
    const v8bf hi = *(const v8bf*)(p + r * ld + h8 + 16);
    v16bf v;
#pragma unroll
    for (int i = 0; i < 8; ++i) { v[i] = lo[i]; v[i + 8] = hi[i]; }
    return v;
}

__device__ inline v8bf cvt8(const float4 a, const float4 b) {
    v8bf r;
    r[0] = (__bf16)a.x; r[1] = (__bf16)a.y; r[2] = (__bf16)a.z; r[3] = (__bf16)a.w;
    r[4] = (__bf16)b.x; r[5] = (__bf16)b.y; r[6] = (__bf16)b.z; r[7] = (__bf16)b.w;
    return r;
}

// ---------------------------------------------------------------------------
// Prep: elementwise f32 -> bf16 (8 elems/thread, b128 in / b128 out).
// ---------------------------------------------------------------------------
__global__ __launch_bounds__(256) void cvt_to_bf16(const float* __restrict__ x,
                                                   __bf16* __restrict__ y) {
    const size_t i = ((size_t)blockIdx.x * 256 + threadIdx.x) * 8;
    const float4* g = (const float4*)(x + i);
    *(v8bf*)(y + i) = cvt8(g[0], g[1]);
}

// Prep: W (1024x1024 f32, K-major) -> Wt (1024x1024 bf16, N-major).
__global__ __launch_bounds__(256) void transpose_w(const float* __restrict__ W,
                                                   __bf16* __restrict__ Wt) {
    __shared__ float t[64 * 65];
    const int r0 = blockIdx.y * 64, c0 = blockIdx.x * 64;
    for (int i = threadIdx.x; i < 64 * 64; i += 256) {
        const int r = i >> 6, c = i & 63;
        t[r * 65 + c] = W[(size_t)(r0 + r) * D_MODEL + c0 + c];
    }
    __syncthreads();
    for (int i = threadIdx.x; i < 64 * 64; i += 256) {
        const int c = i >> 6, r = i & 63;
        Wt[(size_t)(c0 + c) * D_MODEL + r0 + r] = (__bf16)t[r * 65 + c];
    }
}

// ---------------------------------------------------------------------------
// GEMM: C[4096x1024] = X(bf16) @ Wt(bf16,N-major)^T + bias.
// Tile 128x128, K-step 32; TDM double-buffered; one barrier per k-step.
// mode 0: head-split bf16 (B,H,S,64); mode 1: plain f32 (B,S,D);
// mode 2: per-head transposed bf16 (B,H,64,S).
// ---------------------------------------------------------------------------
__global__ __launch_bounds__(256) void gemm_bf16(const __bf16* __restrict__ X,
                                                 const __bf16* __restrict__ Wt,
                                                 const float* __restrict__ bias,
                                                 void* __restrict__ Yv, int mode) {
    __shared__ __attribute__((aligned(16))) __bf16 sA[2][128 * LDL];
    __shared__ __attribute__((aligned(16))) __bf16 sB[2][128 * LDL];

    const int tid  = threadIdx.x;
    const int lane = tid & 31;
    const int wave = tid >> 5;
    const int m0   = blockIdx.y * 128;
    const int n0   = blockIdx.x * 128;
    const int wm   = (wave & 3) * 32;
    const int wn   = (wave >> 2) * 64;

    v8f acc[2][4] = {};

    if (wave == 0) {
        tdm_load_2d(lds_off(sA[0]), X  + (size_t)m0 * D_MODEL, 32, 128, D_MODEL, 3, 3);
        tdm_load_2d(lds_off(sB[0]), Wt + (size_t)n0 * D_MODEL, 32, 128, D_MODEL, 3, 3);
    }

    for (int t = 0; t < D_MODEL / 32; ++t) {
        const int cur = t & 1;
        if (wave == 0) __builtin_amdgcn_s_wait_tensorcnt(0);
        __syncthreads();   // tile t landed; all prior reads of buffer cur^1 done
        if (wave == 0 && t + 1 < D_MODEL / 32) {
            const int k1 = (t + 1) * 32;
            tdm_load_2d(lds_off(sA[cur ^ 1]), X  + (size_t)m0 * D_MODEL + k1, 32, 128, D_MODEL, 3, 3);
            tdm_load_2d(lds_off(sB[cur ^ 1]), Wt + (size_t)n0 * D_MODEL + k1, 32, 128, D_MODEL, 3, 3);
        }
        v16bf af[2], bf[4];
#pragma unroll
        for (int fm = 0; fm < 2; ++fm) af[fm] = frag16(sA[cur] + (wm + fm * 16) * LDL, lane, LDL);
#pragma unroll
        for (int fn = 0; fn < 4; ++fn) bf[fn] = frag16(sB[cur] + (wn + fn * 16) * LDL, lane, LDL);
#pragma unroll
        for (int fm = 0; fm < 2; ++fm)
#pragma unroll
            for (int fn = 0; fn < 4; ++fn)
                acc[fm][fn] = __builtin_amdgcn_wmma_f32_16x16x32_bf16(
                    false, af[fm], false, bf[fn], (short)0, acc[fm][fn], false, false);
    }

    const int col0 = lane & 15;
    const int rb   = (lane >> 4) * 8;
#pragma unroll
    for (int fm = 0; fm < 2; ++fm)
#pragma unroll
        for (int fn = 0; fn < 4; ++fn)
#pragma unroll
            for (int r = 0; r < 8; ++r) {
                const int row = m0 + wm + fm * 16 + rb + r;
                const int col = n0 + wn + fn * 16 + col0;
                const float val = acc[fm][fn][r] + bias[col];
                const int b = row >> 11, s = row & (S_LEN - 1);
                const int h = col >> 6,  dd = col & (DEPTH - 1);
                if (mode == 1) {
                    ((float*)Yv)[(size_t)row * D_MODEL + col] = val;
                } else if (mode == 0) {
                    ((__bf16*)Yv)[(((size_t)b * NHEAD + h) * S_LEN + s) * DEPTH + dd] = (__bf16)val;
                } else {  // mode 2: V^T per head (B,H,DEPTH,S)
                    ((__bf16*)Yv)[(((size_t)b * NHEAD + h) * DEPTH + dd) * S_LEN + s] = (__bf16)val;
                }
            }
}

// ---------------------------------------------------------------------------
// Logits: per (b,h)  attn = 0.125 * Qh Kh^T + mask.  Full-depth (K=64) tiles
// loaded once via TDM (128x64, LDS stride 72 elems), 16 WMMAs, one barrier.
// ---------------------------------------------------------------------------
__global__ __launch_bounds__(256) void attn_logits(const __bf16* __restrict__ qh,
                                                   const __bf16* __restrict__ kh,
                                                   const float* __restrict__ mask,
                                                   float* __restrict__ attn) {
    __shared__ __attribute__((aligned(16))) __bf16 sQ[128 * LDQ];
    __shared__ __attribute__((aligned(16))) __bf16 sK[128 * LDQ];

    const int bh   = blockIdx.z;
    const int b    = bh >> 4;
    const int tid  = threadIdx.x;
    const int lane = tid & 31;
    const int wave = tid >> 5;
    const int m0   = blockIdx.y * 128;
    const int n0   = blockIdx.x * 128;
    const int wm   = (wave & 3) * 32;
    const int wn   = (wave >> 2) * 64;

    const __bf16* Q = qh + (size_t)bh * S_LEN * DEPTH;
    const __bf16* K = kh + (size_t)bh * S_LEN * DEPTH;

    if (wave == 0) {
        tdm_load_2d(lds_off(sQ), Q + (size_t)m0 * DEPTH, 64, 128, DEPTH, 4, 3);
        tdm_load_2d(lds_off(sK), K + (size_t)n0 * DEPTH, 64, 128, DEPTH, 4, 3);
        __builtin_amdgcn_s_wait_tensorcnt(0);
    }
    __syncthreads();

    v8f acc[2][4] = {};
#pragma unroll
    for (int k0 = 0; k0 < DEPTH; k0 += 32) {
        v16bf af[2], bf[4];
#pragma unroll
        for (int fm = 0; fm < 2; ++fm) af[fm] = frag16(sQ + (wm + fm * 16) * LDQ + k0, lane, LDQ);
#pragma unroll
        for (int fn = 0; fn < 4; ++fn) bf[fn] = frag16(sK + (wn + fn * 16) * LDQ + k0, lane, LDQ);
#pragma unroll
        for (int fm = 0; fm < 2; ++fm)
#pragma unroll
            for (int fn = 0; fn < 4; ++fn)
                acc[fm][fn] = __builtin_amdgcn_wmma_f32_16x16x32_bf16(
                    false, af[fm], false, bf[fn], (short)0, acc[fm][fn], false, false);
    }

    const int col0 = lane & 15;
    const int rb   = (lane >> 4) * 8;
#pragma unroll
    for (int fm = 0; fm < 2; ++fm)
#pragma unroll
        for (int fn = 0; fn < 4; ++fn)
#pragma unroll
            for (int r = 0; r < 8; ++r) {
                const int row = m0 + wm + fm * 16 + rb + r;     // query index
                const int col = n0 + wn + fn * 16 + col0;       // key index
                const float m = fmaxf(mask[(size_t)b * S_LEN + col],
                                      (col > row) ? 1.0f : 0.0f);
                attn[((size_t)bh * S_LEN + row) * S_LEN + col] =
                    acc[fm][fn][r] * 0.125f + m * (-1e17f);
            }
}

// ---------------------------------------------------------------------------
// In-place softmax over each attn row (2048 f32). 256 thr/row, 8 elems/thr.
// ---------------------------------------------------------------------------
__global__ __launch_bounds__(256) void softmax_rows(float* __restrict__ attn) {
    float* p = attn + (size_t)blockIdx.x * S_LEN;
    const int tid  = threadIdx.x;
    const int lane = tid & 31;
    const int wave = tid >> 5;
    __shared__ float redmax[8];
    __shared__ float redsum[8];

    const float4 a = ((const float4*)p)[tid * 2 + 0];
    const float4 c = ((const float4*)p)[tid * 2 + 1];
    float x[8] = {a.x, a.y, a.z, a.w, c.x, c.y, c.z, c.w};

    float m = x[0];
#pragma unroll
    for (int i = 1; i < 8; ++i) m = fmaxf(m, x[i]);
#pragma unroll
    for (int off = 16; off > 0; off >>= 1) m = fmaxf(m, __shfl_xor(m, off, 32));
    if (lane == 0) redmax[wave] = m;
    __syncthreads();
    float gmax = redmax[0];
#pragma unroll
    for (int w = 1; w < 8; ++w) gmax = fmaxf(gmax, redmax[w]);

    float s = 0.0f;
#pragma unroll
    for (int i = 0; i < 8; ++i) { x[i] = __expf(x[i] - gmax); s += x[i]; }
#pragma unroll
    for (int off = 16; off > 0; off >>= 1) s += __shfl_xor(s, off, 32);
    if (lane == 0) redsum[wave] = s;
    __syncthreads();
    float gsum = 0.0f;
#pragma unroll
    for (int w = 0; w < 8; ++w) gsum += redsum[w];
    const float inv = 1.0f / gsum;

    const float4 o0 = {x[0] * inv, x[1] * inv, x[2] * inv, x[3] * inv};
    const float4 o1 = {x[4] * inv, x[5] * inv, x[6] * inv, x[7] * inv};
    ((float4*)p)[tid * 2 + 0] = o0;
    ((float4*)p)[tid * 2 + 1] = o1;
}

// ---------------------------------------------------------------------------
// attn @ V: per (b,h)  ctx = attn(2048x2048 f32) @ Vh(2048x64).
// A tile: float4 loads + packed cvt + ds_store_b128 (f32 source is mandated).
// B tile: TDM from Vh^T (B,H,64,S), double-buffered on TENSORcnt.
// ---------------------------------------------------------------------------
__global__ __launch_bounds__(256) void attn_v(const float* __restrict__ attn,
                                              const __bf16* __restrict__ vhT,
                                              __bf16* __restrict__ ctx) {
    __shared__ __attribute__((aligned(16))) __bf16 sA[128 * LDL];
    __shared__ __attribute__((aligned(16))) __bf16 sB[2][64 * LDL];

    const int bh   = blockIdx.z;
    const int b    = bh >> 4, h = bh & 15;
    const int tid  = threadIdx.x;
    const int lane = tid & 31;
    const int wave = tid >> 5;
    const int m0   = blockIdx.x * 128;
    const int wm   = (wave & 3) * 32;
    const int wn   = (wave >> 2) * 32;

    const float*  A = attn + (size_t)bh * S_LEN * S_LEN;
    const __bf16* V = vhT  + (size_t)bh * DEPTH * S_LEN;   // rows d, stride S

    v8f acc[2][2] = {};

    if (wave == 0)
        tdm_load_2d(lds_off(sB[0]), V, 32, 64, S_LEN, 3, 3);

    for (int t = 0; t < S_LEN / 32; ++t) {
        const int k0 = t * 32, cur = t & 1;
        // Stage attn tile 128x32 f32 -> bf16 (8 elems per store).
#pragma unroll
        for (int i = tid; i < 512; i += 256) {
            const int r = i >> 2, c8 = (i & 3) * 8;
            const float4* g = (const float4*)(A + (size_t)(m0 + r) * S_LEN + k0 + c8);
            *(v8bf*)(sA + r * LDL + c8) = cvt8(g[0], g[1]);
        }
        if (wave == 0) __builtin_amdgcn_s_wait_tensorcnt(0);
        __syncthreads();
        if (wave == 0 && t + 1 < S_LEN / 32)
            tdm_load_2d(lds_off(sB[cur ^ 1]), V + (t + 1) * 32, 32, 64, S_LEN, 3, 3);

        v16bf af[2], bf[2];
#pragma unroll
        for (int fm = 0; fm < 2; ++fm) af[fm] = frag16(sA + (wm + fm * 16) * LDL, lane, LDL);
#pragma unroll
        for (int fn = 0; fn < 2; ++fn) bf[fn] = frag16(sB[cur] + (wn + fn * 16) * LDL, lane, LDL);
#pragma unroll
        for (int fm = 0; fm < 2; ++fm)
#pragma unroll
            for (int fn = 0; fn < 2; ++fn)
                acc[fm][fn] = __builtin_amdgcn_wmma_f32_16x16x32_bf16(
                    false, af[fm], false, bf[fn], (short)0, acc[fm][fn], false, false);
        __syncthreads();   // protect sA restage next iteration
    }

    const int col0 = lane & 15;
    const int rb   = (lane >> 4) * 8;
#pragma unroll
    for (int fm = 0; fm < 2; ++fm)
#pragma unroll
        for (int fn = 0; fn < 2; ++fn)
#pragma unroll
            for (int r = 0; r < 8; ++r) {
                const int qrow = m0 + wm + fm * 16 + rb + r;
                const int d    = wn + fn * 16 + col0;
                ctx[((size_t)b * S_LEN + qrow) * D_MODEL + h * DEPTH + d] =
                    (__bf16)acc[fm][fn][r];
            }
}

// ---------------------------------------------------------------------------
// Launch. Inputs: 0:v 1:k 2:q 3:mask 4:Wq 5:bq 6:Wk 7:bk 8:Wv 9:bv 10:Wo 11:bo
// d_out = [out (B,S,D) f32 | attn (B,H,S,S) f32]
// ws (bf16): xq|xk|xv (3x4M) | WqT|WkT|WvT|WoT (4x1M) | qh|kh|vhT|ctx (4x4M)
//   total 32M bf16 = 64 MB.
// ---------------------------------------------------------------------------
extern "C" void kernel_launch(void* const* d_in, const int* in_sizes, int n_in,
                              void* d_out, int out_size, void* d_ws, size_t ws_size,
                              hipStream_t stream) {
    (void)in_sizes; (void)n_in; (void)out_size; (void)ws_size;
    const float* vin  = (const float*)d_in[0];
    const float* kin  = (const float*)d_in[1];
    const float* qin  = (const float*)d_in[2];
    const float* mask = (const float*)d_in[3];
    const float* Wq   = (const float*)d_in[4];
    const float* bq   = (const float*)d_in[5];
    const float* Wk   = (const float*)d_in[6];
    const float* bk   = (const float*)d_in[7];
    const float* Wv   = (const float*)d_in[8];
    const float* bv   = (const float*)d_in[9];
    const float* Wo   = (const float*)d_in[10];
    const float* bo   = (const float*)d_in[11];

    float* out  = (float*)d_out;
    float* attn = out + (size_t)BATCH * S_LEN * D_MODEL;

    const size_t TM = (size_t)MROWS * D_MODEL;     // 4,194,304
    const size_t TW = (size_t)D_MODEL * D_MODEL;   // 1,048,576
    __bf16* w   = (__bf16*)d_ws;
    __bf16 *xq = w,        *xk = xq + TM,  *xv = xk + TM;
    __bf16 *WqT = xv + TM, *WkT = WqT + TW, *WvT = WkT + TW, *WoT = WvT + TW;
    __bf16 *qh = WoT + TW, *kh = qh + TM,  *vhT = kh + TM,  *ctx = vhT + TM;

    const dim3 blk(256);
    const unsigned gcv = (unsigned)(TM / (256 * 8));           // 2048
    cvt_to_bf16<<<gcv, blk, 0, stream>>>(qin, xq);
    cvt_to_bf16<<<gcv, blk, 0, stream>>>(kin, xk);
    cvt_to_bf16<<<gcv, blk, 0, stream>>>(vin, xv);

    const dim3 gtr(16, 16);
    transpose_w<<<gtr, blk, 0, stream>>>(Wq, WqT);
    transpose_w<<<gtr, blk, 0, stream>>>(Wk, WkT);
    transpose_w<<<gtr, blk, 0, stream>>>(Wv, WvT);
    transpose_w<<<gtr, blk, 0, stream>>>(Wo, WoT);

    const dim3 gproj(D_MODEL / 128, MROWS / 128);              // 8 x 32
    gemm_bf16<<<gproj, blk, 0, stream>>>(xq, WqT, bq, qh,  0);
    gemm_bf16<<<gproj, blk, 0, stream>>>(xk, WkT, bk, kh,  0);
    gemm_bf16<<<gproj, blk, 0, stream>>>(xv, WvT, bv, vhT, 2);

    const dim3 glog(S_LEN / 128, S_LEN / 128, BATCH * NHEAD);  // 16 x 16 x 32
    attn_logits<<<glog, blk, 0, stream>>>(qh, kh, mask, attn);

    softmax_rows<<<dim3(BATCH * NHEAD * S_LEN), blk, 0, stream>>>(attn);

    const dim3 gav(S_LEN / 128, 1, BATCH * NHEAD);             // 16 x 1 x 32
    attn_v<<<gav, blk, 0, stream>>>(attn, vhT, ctx);

    gemm_bf16<<<gproj, blk, 0, stream>>>(ctx, WoT, bo, out, 1);
}
